// Soft_KMeansCriterion1_2138893713645
// MI455X (gfx1250) — matrix operand order repaired and checked by reading, not compiled, not run on hardware
//
#include <hip/hip_runtime.h>
#include <hip/hip_bf16.h>

#define BETA_F 10.0f
#define NROWS 131072
#define NDIM 128
#define NCENT 512
#define WAVES_PER_BLOCK 4
#define ROWS_PER_BLOCK 64              // 4 waves * 16 rows
#define NBLK (NROWS / ROWS_PER_BLOCK)  // 2048
#define DSTRIDE 516                    // padded row stride (floats) for LDS dist slab
#define NFRAG_ELEMS (NCENT * NDIM)     // 65536 bf16 elements per plane

typedef __attribute__((ext_vector_type(16))) __bf16 v16bf;
typedef __attribute__((ext_vector_type(16))) unsigned short v16us;
typedef __attribute__((ext_vector_type(8))) float v8f;

static __device__ __forceinline__ unsigned short f2bf_rn(float x) {
  unsigned u = __float_as_uint(x);
  u += 0x7FFFu + ((u >> 16) & 1u);
  return (unsigned short)(u >> 16);
}
static __device__ __forceinline__ float bf2f(unsigned short h) {
  return __uint_as_float(((unsigned)h) << 16);
}

// ---------------------------------------------------------------------------
// Kernel 1a: centroid squared norms (512 values, fp32-exact) -> workspace
// ---------------------------------------------------------------------------
__global__ __launch_bounds__(128) void c2_kernel(const float* __restrict__ cen,
                                                 float* __restrict__ c2g) {
  int i = blockIdx.x * 128 + threadIdx.x;
  if (i < NCENT) {
    const float* r = cen + (size_t)i * NDIM;
    float s = 0.f;
#pragma unroll 8
    for (int d = 0; d < NDIM; ++d) s += r[d] * r[d];
    c2g[i] = s;
  }
}

// ---------------------------------------------------------------------------
// Kernel 1b: ONE-TIME centroid fp32 -> compensated bf16 (hi,lo), scattered
// into the exact WMMA B-fragment layout so the main kernel does pure loads:
//   fragment f = (t*4 + kc)*32 + lane ; element j -> K = kc*32 + (j/8)*16 + (lane/16)*8 + (j%8)
//   centroid n = t*16 + (lane%16)
// ---------------------------------------------------------------------------
__global__ __launch_bounds__(256)
void bprep_kernel(const float* __restrict__ cen,
                  unsigned short* __restrict__ bhi,
                  unsigned short* __restrict__ blo) {
  int idx = blockIdx.x * 256 + threadIdx.x;  // 0 .. 65535
  if (idx >= NFRAG_ELEMS) return;
  int n = idx >> 7;          // centroid 0..511
  int k = idx & 127;         // feature 0..127
  float x = cen[(size_t)n * NDIM + k];
  unsigned short hs = f2bf_rn(x);
  unsigned short ls = f2bf_rn(x - bf2f(hs));
  int t = n >> 4, l16 = n & 15;
  int kc = k >> 5, rem = k & 31;
  int half = (rem >> 3) & 1, j = (rem & 7) | ((rem >> 4) << 3);
  int lane = half * 16 + l16;
  size_t off = ((size_t)((t * 4 + kc) * 32 + lane) << 4) + j;
  bhi[off] = hs;
  blo[off] = ls;
}

// ---------------------------------------------------------------------------
// Kernel 2: main — compensated-bf16 WMMA GEMM, stable softmax, coalesced
// assignment stores, per-block loss partials. 128 threads = 4 wave32.
// ---------------------------------------------------------------------------
__global__ __launch_bounds__(128)
void soft_kmeans_main(const float* __restrict__ emb,
                      const unsigned short* __restrict__ bhig,
                      const unsigned short* __restrict__ blog,
                      const float* __restrict__ c2g, float* __restrict__ out,
                      float* __restrict__ partial) {
  __shared__ float c2s[NCENT];
  __shared__ float dists[WAVES_PER_BLOCK][16][DSTRIDE];
  __shared__ float wloss[WAVES_PER_BLOCK];

  const int tid  = threadIdx.x;
  const int wave = tid >> 5;
  const int lane = tid & 31;
  const int half = lane >> 4;  // 0 or 1
  const int l16  = lane & 15;

  for (int i = tid; i < NCENT; i += 128) c2s[i] = c2g[i];
  __syncthreads();

  const int r0  = blockIdx.x * ROWS_PER_BLOCK + wave * 16;
  const int row = r0 + l16;
  const float* erow = emb + (size_t)row * NDIM;

  // ---- A fragments (bf16 hi/lo split), ISA 16-bit A 16x32 layout:
  // lane half h, element j<8 : K = kc*32 + h*8 + j ; j>=8: K = kc*32 + 16 + h*8 + (j-8)
  v16us ahiu[4], alou[4];
  float e2 = 0.f;
#pragma unroll
  for (int kc = 0; kc < 4; ++kc) {
    const int kb = kc * 32 + half * 8;
#pragma unroll
    for (int j = 0; j < 8; ++j) {
      float x = erow[kb + j];
      unsigned short hs = f2bf_rn(x);
      ahiu[kc][j] = hs;
      alou[kc][j] = f2bf_rn(x - bf2f(hs));
      e2 += x * x;
    }
#pragma unroll
    for (int j = 0; j < 8; ++j) {
      float x = erow[kb + 16 + j];
      unsigned short hs = f2bf_rn(x);
      ahiu[kc][8 + j] = hs;
      alou[kc][8 + j] = f2bf_rn(x - bf2f(hs));
      e2 += x * x;
    }
  }
  e2 += __shfl_xor(e2, 16, 32);  // lane pair covered the full 128-d row
  float e2r[8];
#pragma unroll
  for (int v = 0; v < 8; ++v) e2r[v] = __shfl(e2, v + 8 * half, 32);

  // pre-formatted B fragment planes (32B contiguous per lane per (t,kc))
  const v16us* bhf = (const v16us*)bhig;
  const v16us* blf = (const v16us*)blog;

  // ---- Column-tile loop: 32 tiles of 16 centroids
  for (int t = 0; t < 32; ++t) {
    v8f acc = {};
    const int fbase = (t * 4) * 32 + lane;
#pragma unroll
    for (int kc = 0; kc < 4; ++kc) {
      v16us bhiu = bhf[fbase + kc * 32];
      v16us blou = blf[fbase + kc * 32];
      v16bf Ah = __builtin_bit_cast(v16bf, ahiu[kc]);
      v16bf Al = __builtin_bit_cast(v16bf, alou[kc]);
      v16bf Bh = __builtin_bit_cast(v16bf, bhiu);
      v16bf Bl = __builtin_bit_cast(v16bf, blou);
      // cross ~= Ah*Bh + Al*Bh + Ah*Bl  (fp32-accurate compensated bf16)
      acc = __builtin_amdgcn_wmma_f32_16x16x32_bf16(false, Ah, false, Bh,
                                                    (short)0, acc, false, false);
      acc = __builtin_amdgcn_wmma_f32_16x16x32_bf16(false, Al, false, Bh,
                                                    (short)0, acc, false, false);
      acc = __builtin_amdgcn_wmma_f32_16x16x32_bf16(false, Ah, false, Bl,
                                                    (short)0, acc, false, false);
    }
    // C/D layout: lane col n = l16, element v -> row v + 8*half
    const float c2v = c2s[t * 16 + l16];
#pragma unroll
    for (int v = 0; v < 8; ++v) {
      float sq = e2r[v] + c2v - 2.0f * acc[v];
      sq = fmaxf(sq, 0.0f);
      dists[wave][v + 8 * half][t * 16 + l16] = __builtin_sqrtf(sq);
    }
  }

  // ---- Stable softmax of -beta*d : row MIN (jax softmax re-stabilizes)
  const int cbase = half * 256;
  const float4* drow = (const float4*)&dists[wave][l16][cbase];
  float m = 3.4e38f;
#pragma unroll 4
  for (int i = 0; i < 64; ++i) {
    float4 d4 = drow[i];
    m = fminf(m, fminf(fminf(d4.x, d4.y), fminf(d4.z, d4.w)));
  }
  m = fminf(m, __shfl_xor(m, 16, 32));
  float s = 0.f;
#pragma unroll 4
  for (int i = 0; i < 64; ++i) {
    float4 d4 = drow[i];
    s += __expf(-BETA_F * (d4.x - m)) + __expf(-BETA_F * (d4.y - m)) +
         __expf(-BETA_F * (d4.z - m)) + __expf(-BETA_F * (d4.w - m));
  }
  s += __shfl_xor(s, 16, 32);

  // ---- Write assignments (coalesced 128B/instr) + loss partial
  float lossp = 0.f;
  float* outA = out + 1;  // d_out[0] is the loss scalar
  for (int r = 0; r < 16; ++r) {
    const float mr  = __shfl(m, r, 32);
    const float inv = 1.0f / __shfl(s, r, 32);
    const size_t gbase = (size_t)(r0 + r) * NCENT;
#pragma unroll
    for (int it = 0; it < 16; ++it) {
      const int c = it * 32 + lane;
      const float d = dists[wave][r][c];
      const float a = __expf(-BETA_F * (d - mr)) * inv;
      outA[gbase + c] = a;
      lossp += a * d;
    }
  }
#pragma unroll
  for (int off = 16; off >= 1; off >>= 1) lossp += __shfl_xor(lossp, off, 32);
  if (lane == 0) wloss[wave] = lossp;
  __syncthreads();
  if (tid == 0)
    partial[blockIdx.x] = wloss[0] + wloss[1] + wloss[2] + wloss[3];
}

// ---------------------------------------------------------------------------
// Kernel 3: deterministic reduction of 2048 block partials -> loss = sum / N
// ---------------------------------------------------------------------------
__global__ __launch_bounds__(256) void loss_reduce(const float* __restrict__ partial,
                                                   float* __restrict__ out) {
  __shared__ float sm[256];
  float s = 0.f;
  for (int i = threadIdx.x; i < NBLK; i += 256) s += partial[i];
  sm[threadIdx.x] = s;
  __syncthreads();
  for (int off = 128; off >= 1; off >>= 1) {
    if ((int)threadIdx.x < off) sm[threadIdx.x] += sm[threadIdx.x + off];
    __syncthreads();
  }
  if (threadIdx.x == 0) out[0] = sm[0] / (float)NROWS;
}

// ---------------------------------------------------------------------------
extern "C" void kernel_launch(void* const* d_in, const int* in_sizes, int n_in,
                              void* d_out, int out_size, void* d_ws, size_t ws_size,
                              hipStream_t stream) {
  (void)in_sizes; (void)n_in; (void)out_size; (void)ws_size;
  const float* emb = (const float*)d_in[0];  // (131072, 128) fp32
  const float* cen = (const float*)d_in[1];  // (512, 128) fp32
  float* out = (float*)d_out;                // [loss, assignments(131072*512)]
  float* ws  = (float*)d_ws;
  float* partial = ws;                                  // 2048 floats
  float* c2g     = ws + NBLK;                           // 512 floats
  unsigned short* bhi = (unsigned short*)(ws + NBLK + NCENT);  // 32B-aligned
  unsigned short* blo = bhi + NFRAG_ELEMS;

  c2_kernel<<<(NCENT + 127) / 128, 128, 0, stream>>>(cen, c2g);
  bprep_kernel<<<(NFRAG_ELEMS + 255) / 256, 256, 0, stream>>>(cen, bhi, blo);
  soft_kmeans_main<<<NBLK, 128, 0, stream>>>(emb, bhi, blo, c2g, out, partial);
  loss_reduce<<<1, 256, 0, stream>>>(partial, out);
}
